// Attention_67370857005350
// MI455X (gfx1250) — matrix operand order repaired and checked by reading, not compiled, not run on hardware
//
#include <hip/hip_runtime.h>
#include <hip/hip_bf16.h>

typedef __attribute__((ext_vector_type(16))) _Float16     v16h;
typedef __attribute__((ext_vector_type(8)))  _Float16     v8h;
typedef __attribute__((ext_vector_type(8)))  float        v8f;
typedef __attribute__((ext_vector_type(4)))  unsigned int u32x4;
typedef __attribute__((ext_vector_type(8)))  int          i32x8;
typedef __attribute__((ext_vector_type(4)))  int          i32x4;

// --------------------------------------------------------------------------
// TDM availability (ROCm 7.2 clang-22: 5-arg builtin; clang>=23: 6-arg).
// --------------------------------------------------------------------------
#if defined(__HIP_DEVICE_COMPILE__) && defined(__has_builtin)
#if __has_builtin(__builtin_amdgcn_tensor_load_to_lds) && \
    __has_builtin(__builtin_amdgcn_s_wait_tensorcnt)
#define HAVE_TDM 1
#endif
#endif
#ifndef HAVE_TDM
#define HAVE_TDM 0
#endif

// Diagnostics (visible in compile stderr) to confirm which path is live.
#if defined(__HIP_DEVICE_COMPILE__)
#if HAVE_TDM
#pragma message("CDNA5-DIAG: device pass, TDM tensor_load_to_lds path ENABLED")
#if __clang_major__ >= 23
#pragma message("CDNA5-DIAG: using 6-arg tensor_load_to_lds (clang>=23)")
#else
#pragma message("CDNA5-DIAG: using 5-arg tensor_load_to_lds (clang<=22)")
#endif
#else
#pragma message("CDNA5-DIAG: device pass, TDM builtin NOT found -> vector-staging fallback")
#endif
#endif

__device__ __forceinline__ v16h cat8(v8h lo, v8h hi) {
    return __builtin_shufflevector(lo, hi, 0, 1, 2, 3, 4, 5, 6, 7,
                                   8, 9, 10, 11, 12, 13, 14, 15);
}

// Per-lane A-fragment = one contiguous 16-half run of a 32-long K row:
//   halves 0..15 -> k = abase..abase+15
// where abase = (lane&16) ? 16 : 0.
__device__ __forceinline__ v16h load_a_frag_lds(const _Float16* rowp, int lane) {
    const int ab = (lane & 16) ? 16 : 0;
    v8h lo = *(const v8h*)(rowp + ab);
    v8h hi = *(const v8h*)(rowp + ab + 8);
    return cat8(lo, hi);
}

#if HAVE_TDM
// ---------------------------------------------------------------------------
// TDM: DMA a rows x cols f16 tile (row stride in elements) global -> LDS.
// D# per CDNA5 ISA ch.8. Descriptor words made wave-uniform via readfirstlane.
// ---------------------------------------------------------------------------
__device__ __forceinline__ void tdm_load_tile_f16(const _Float16* gsrc,
                                                  unsigned lds_off,
                                                  int cols, int rows,
                                                  int row_stride) {
    unsigned lo   = (unsigned)__builtin_amdgcn_readfirstlane((int)(unsigned)(uintptr_t)gsrc);
    unsigned hi   = (unsigned)__builtin_amdgcn_readfirstlane((int)(unsigned)((uintptr_t)gsrc >> 32));
    unsigned lofs = (unsigned)__builtin_amdgcn_readfirstlane((int)lds_off);
    u32x4 g0;
    g0[0] = 1u;                                     // count=1, user descriptor
    g0[1] = lofs;                                   // lds_addr (bytes)
    g0[2] = lo;                                     // global_addr[31:0]
    g0[3] = (hi & 0x01FFFFFFu) | (2u << 30);        // global_addr[56:32] | type=2
    i32x8 g1;
    g1[0] = 0x00010000;                             // wg_mask=0, data_size=2B
    g1[1] = (row_stride & 0xFFFF) << 16;            // tensor_dim0[15:0]
    g1[2] = ((row_stride >> 16) & 0xFFFF)
          | ((rows & 0xFFFF) << 16);                // tensor_dim0 hi | tensor_dim1 lo
    g1[3] = ((cols & 0xFFFF) << 16);                // tile_dim0
    g1[4] = (rows & 0xFFFF);                        // tile_dim1 (tile_dim2=0)
    g1[5] = row_stride;                             // tensor_dim0_stride[31:0]
    g1[6] = 0;
    g1[7] = 0;
    i32x4 gz = {0, 0, 0, 0};
#if __clang_major__ >= 23
    i32x8 gz8 = {0, 0, 0, 0, 0, 0, 0, 0};
    __builtin_amdgcn_tensor_load_to_lds(g0, g1, gz, gz, gz8, 0);
#else
    __builtin_amdgcn_tensor_load_to_lds(g0, g1, gz, gz, 0);
#endif
}
#endif

// ===========================================================================
// Kernel 1: depthwise 3x3 conv + BN(eval) + ReLU, f32 NCHW in ->
//           f16 pixel-major out [b][n][c]  (c fastest -> coalesced stores)
// ===========================================================================
__global__ void dw_bn_relu_kernel(const float* __restrict__ in,
                                  const float* __restrict__ wdw,
                                  const float* __restrict__ g,
                                  const float* __restrict__ bta,
                                  const float* __restrict__ mu,
                                  const float* __restrict__ var,
                                  _Float16* __restrict__ out,
                                  int C, int H, int W, int Ho, int Wo, int stride) {
    int idx = blockIdx.x * blockDim.x + threadIdx.x;
    int total = 2 * C * Ho * Wo;
    if (idx >= total) return;
    int c = idx % C;
    int t = idx / C;
    int n = t % (Ho * Wo);
    int b = t / (Ho * Wo);
    int y = n / Wo, x = n % Wo;

    const float* ip = in + ((size_t)(b * C + c)) * H * W;
    const float* wp = wdw + c * 9;
    float acc = 0.f;
#pragma unroll
    for (int dy = 0; dy < 3; ++dy) {
#pragma unroll
        for (int dx = 0; dx < 3; ++dx) {
            int iy = y * stride + dy - 1;
            int ix = x * stride + dx - 1;
            if (iy >= 0 && iy < H && ix >= 0 && ix < W)
                acc += ip[iy * W + ix] * wp[dy * 3 + dx];
        }
    }
    float s = g[c] * rsqrtf(var[c] + 1e-5f);
    float v = acc * s + (bta[c] - mu[c] * s);
    out[idx] = (_Float16)fmaxf(v, 0.f);   // layout [b][n][c] == idx
}

// ===========================================================================
// Kernel 2/4: 1x1 conv as LDS-tiled WMMA GEMM over pixel-major activations.
//   X: f16 [2][N][K]; W: f32 [M][K].  Block = 4 waves = 64x64 output tile.
//   All fragments are contiguous v8h/v16h LDS loads (ds_load_b128).
//   mode 0: f16 ReLU, pixel-major out [b][n][m] (vector stores)
//   mode 1: f32 bias+ReLU, channel-major out [b][m][n] (final NCHW output)
//   mode 2: kv split: m<512 -> K pixel-major [b][n][512];
//                     m>=512 -> V channel-major [b][m-512][n]
// ===========================================================================
__global__ void __launch_bounds__(128)
gemm_wmma_kernel(const float* __restrict__ Wt, const _Float16* __restrict__ X,
                 void* __restrict__ out, const float* __restrict__ bias,
                 int M, int N, int K, int mode) {
    __shared__ alignas(32) _Float16 As[64 * 32];   // [m][k]
    __shared__ alignas(32) _Float16 Bs[64 * 32];   // [n][k]  (K innermost!)

    const int lane = threadIdx.x & 31;
    const int wid  = threadIdx.x >> 5;
    const int tilesM = M >> 6, tilesN = N >> 6;
    const int perB = tilesM * tilesN;
    const int b  = blockIdx.x / perB;
    const int tt = blockIdx.x % perB;
    const int M0 = (tt / tilesN) << 6;
    const int N0 = (tt % tilesN) << 6;

    const _Float16* Xb = X + (size_t)b * N * K;
    const int mloc  = lane & 15;
    const int kofb  = (lane & 16) ? 16 : 0;
    const int rbase = (lane & 16) ? 8 : 0;

    v8f acc[4];
#pragma unroll
    for (int q = 0; q < 4; ++q) acc[q] = (v8f){0.f,0.f,0.f,0.f,0.f,0.f,0.f,0.f};

    for (int kb = 0; kb < K; kb += 32) {
        __syncthreads();
        {   // stage A: 64x32, f32 -> f16; 16 contiguous elems / thread
            const int m  = threadIdx.x >> 1;
            const int kh = (threadIdx.x & 1) << 4;
            const float* src = Wt + (size_t)(M0 + m) * K + kb + kh;
            v16h pack;
#pragma unroll
            for (int i = 0; i < 16; ++i) pack[i] = (_Float16)src[i];
            *(v16h*)&As[m * 32 + kh] = pack;
        }
        {   // stage B: 64 pixels x 32 ch; one 32B vector copy / thread
            const int p  = threadIdx.x >> 1;
            const int kh = (threadIdx.x & 1) << 4;
            *(v16h*)&Bs[p * 32 + kh] =
                *(const v16h*)&Xb[(size_t)(N0 + p) * K + kb + kh];
        }
        __syncthreads();

        v16h af = load_a_frag_lds(&As[(wid * 16 + mloc) * 32], lane);
#pragma unroll
        for (int q = 0; q < 4; ++q) {
            v16h bf = *(const v16h*)&Bs[(q * 16 + mloc) * 32 + kofb];
            acc[q] = __builtin_amdgcn_wmma_f32_16x16x32_f16(
                false, af, false, bf, (short)0, acc[q], false, false);
        }
    }

#pragma unroll
    for (int q = 0; q < 4; ++q) {
        const int n     = N0 + q * 16 + mloc;
        const int mbase = M0 + wid * 16 + rbase;
        if (mode == 0) {            // pixel-major f16, vector store
            _Float16* O = (_Float16*)out + (size_t)b * N * M;
            v8h pk;
#pragma unroll
            for (int r = 0; r < 8; ++r) pk[r] = (_Float16)fmaxf(acc[q][r], 0.f);
            *(v8h*)&O[(size_t)n * M + mbase] = pk;
        } else if (mode == 1) {     // channel-major f32 + bias + ReLU
            float* O = (float*)out + (size_t)b * M * N;
#pragma unroll
            for (int r = 0; r < 8; ++r) {
                float v = acc[q][r] + bias[mbase + r];
                O[(size_t)(mbase + r) * N + n] = fmaxf(v, 0.f);
            }
        } else {                    // kv split
            if (M0 < 512) {         // K: pixel-major [b][n][512]
                _Float16* Okp = (_Float16*)out + (size_t)b * 1024 * 512;
                v8h pk;
#pragma unroll
                for (int r = 0; r < 8; ++r) pk[r] = (_Float16)fmaxf(acc[q][r], 0.f);
                *(v8h*)&Okp[(size_t)n * 512 + mbase] = pk;
            } else {                // V: channel-major [b][m-512][n]
                _Float16* Ov = (_Float16*)out + (size_t)2 * 1024 * 512
                             + (size_t)b * 512 * 1024;
#pragma unroll
                for (int r = 0; r < 8; ++r)
                    Ov[(size_t)(mbase + r - 512) * 1024 + n] =
                        (_Float16)fmaxf(acc[q][r], 0.f);
            }
        }
    }
}

// ===========================================================================
// Kernel 3: flash-attention.  Nq=4096, Nkv=1024, d=64, heads=8, batch=2.
//   Q pixel-major [b][nq][512]; K pixel-major [b][nkv][512];
//   V channel-major [b][d-ch][nkv].  One wave per 16-row Q tile.
//   K/V chunk tiles double-buffered in LDS via TDM (tensor_load_to_lds +
//   s_wait_tensorcnt) with vectorized fallback.  All WMMA fragments are
//   contiguous v8h/v16h loads.
// ===========================================================================
__device__ __forceinline__ float half16_max(float v) {
#pragma unroll
    for (int m = 1; m <= 8; m <<= 1) v = fmaxf(v, __shfl_xor(v, m, 32));
    return v;
}
__device__ __forceinline__ float half16_sum(float v) {
#pragma unroll
    for (int m = 1; m <= 8; m <<= 1) v += __shfl_xor(v, m, 32);
    return v;
}

__global__ void __launch_bounds__(64)
attention_wmma_kernel(const _Float16* __restrict__ qpm,   // [2][4096][512]
                      const _Float16* __restrict__ kvb,   // K pm + V cm
                      _Float16* __restrict__ abuf) {      // [2][4096][512]
    constexpr int NQ = 4096, NKV = 1024;
    __shared__ alignas(32) _Float16 kls[2][2][32 * 64];   // [wave][buf][nkv32][d64]
    __shared__ alignas(32) _Float16 vls[2][2][64 * 32];   // [wave][buf][d64][nkv32]
    __shared__ alignas(32) _Float16 pls[2][16 * 32];

    const int lane = threadIdx.x & 31;
    const int wid  = threadIdx.x >> 5;
    const int gw   = blockIdx.x * 2 + wid;
    if (gw >= 2 * 8 * 256) return;                 // wave-uniform

    const int b  = gw / (8 * 256);
    const int r0 = gw % (8 * 256);
    const int hh = r0 / 256;
    const int q0 = (r0 % 256) << 4;

    const _Float16* kpm = kvb + (size_t)b * 1024 * 512 + hh * 64;         // [nkv][512]
    const _Float16* vcm = kvb + (size_t)2 * 1024 * 512
                        + ((size_t)b * 512 + hh * 64) * NKV;              // [d][nkv]
    _Float16* pb = pls[wid];

    const int mloc  = lane & 15;
    const int kofb  = (lane & 16) ? 16 : 0;
    const int abse  = (lane & 16) ? 8 : 0;
    const int rbase = (lane & 16) ? 8 : 0;

    // Q A-fragments from pixel-major Q: four contiguous 16B global loads,
    // softmax scale 1/sqrt(64) folded in.
    v16h qa0, qa1;
    {
        const _Float16* qrow = qpm + ((size_t)b * NQ + q0 + mloc) * 512 + hh * 64;
        qa0 = cat8(*(const v8h*)(qrow + kofb), *(const v8h*)(qrow + kofb + 8));
        qa1 = cat8(*(const v8h*)(qrow + 32 + kofb), *(const v8h*)(qrow + 40 + kofb));
        const _Float16 sc = (_Float16)0.125f;
#pragma unroll
        for (int i = 0; i < 16; ++i) { qa0[i] *= sc; qa1[i] *= sc; }
    }

    v8f o[4];
    float mrow[8], lrow[8];
#pragma unroll
    for (int t = 0; t < 4; ++t) o[t] = (v8f){0.f,0.f,0.f,0.f,0.f,0.f,0.f,0.f};
#pragma unroll
    for (int r = 0; r < 8; ++r) { mrow[r] = -1e30f; lrow[r] = 0.f; }

#if HAVE_TDM
    // prologue: DMA chunk 0 (K tile: 32 nkv-rows x 64 d ; V tile: 64 d-rows x 32 nkv)
    tdm_load_tile_f16(kpm, (unsigned)(uintptr_t)&kls[wid][0][0], 64, 32, 512);
    tdm_load_tile_f16(vcm, (unsigned)(uintptr_t)&vls[wid][0][0], 32, 64, NKV);
#endif

    for (int jc = 0; jc < 32; ++jc) {
        const int j0 = jc * 32;
        const _Float16* kt;
        const _Float16* vt;
#if HAVE_TDM
        const int cur = jc & 1;
        kt = &kls[wid][cur][0];
        vt = &vls[wid][cur][0];
        if (jc + 1 < 32) {   // issue-ahead next chunk, then wait for current
            tdm_load_tile_f16(kpm + (size_t)(j0 + 32) * 512,
                              (unsigned)(uintptr_t)&kls[wid][cur ^ 1][0], 64, 32, 512);
            tdm_load_tile_f16(vcm + (j0 + 32),
                              (unsigned)(uintptr_t)&vls[wid][cur ^ 1][0], 32, 64, NKV);
            __builtin_amdgcn_s_wait_tensorcnt(2);
        } else {
            __builtin_amdgcn_s_wait_tensorcnt(0);
        }
#else
        // fallback: wave-cooperative vectorized staging of this chunk
        kt = &kls[wid][0][0];
        vt = &vls[wid][0][0];
        {   // K: 32 rows of 64 halves; lane copies one row (4 x 32B)
            const _Float16* src = kpm + (size_t)(j0 + lane) * 512;
            _Float16* dst = &kls[wid][0][lane * 64];
#pragma unroll
            for (int i = 0; i < 4; ++i)
                *(v16h*)&dst[i * 16] = *(const v16h*)&src[i * 16];
        }
#pragma unroll
        for (int rr = 0; rr < 2; ++rr) {   // V: 64 rows of 32 halves; 2 rows/lane
            const int row = lane + rr * 32;
            *(v16h*)&vls[wid][0][row * 32] =
                *(const v16h*)&vcm[(size_t)row * NKV + j0];
            *(v16h*)&vls[wid][0][row * 32 + 16] =
                *(const v16h*)&vcm[(size_t)row * NKV + j0 + 16];
        }
        asm volatile("s_wait_dscnt 0x0" ::: "memory");
#endif

        // ---- S = (Q*scale) K^T : B-frag (k=d, n=kv) = contiguous d-run ----
        v8f s0 = (v8f){0.f,0.f,0.f,0.f,0.f,0.f,0.f,0.f};
        v8f s1 = s0;
        v16h bf;
        bf = *(const v16h*)&kt[mloc * 64 + kofb];
        s0 = __builtin_amdgcn_wmma_f32_16x16x32_f16(false, qa0, false, bf, (short)0, s0, false, false);
        bf = *(const v16h*)&kt[mloc * 64 + 32 + kofb];
        s0 = __builtin_amdgcn_wmma_f32_16x16x32_f16(false, qa1, false, bf, (short)0, s0, false, false);
        bf = *(const v16h*)&kt[(16 + mloc) * 64 + kofb];
        s1 = __builtin_amdgcn_wmma_f32_16x16x32_f16(false, qa0, false, bf, (short)0, s1, false, false);
        bf = *(const v16h*)&kt[(16 + mloc) * 64 + 32 + kofb];
        s1 = __builtin_amdgcn_wmma_f32_16x16x32_f16(false, qa1, false, bf, (short)0, s1, false, false);

        // ---- online softmax over this 32-column block ----
#pragma unroll
        for (int r = 0; r < 8; ++r) {
            float bm = half16_max(fmaxf(s0[r], s1[r]));
            float mnew = fmaxf(mrow[r], bm);
            float alpha = __expf(mrow[r] - mnew);
            float p0 = __expf(s0[r] - mnew);
            float p1 = __expf(s1[r] - mnew);
            float bs = half16_sum(p0 + p1);
            lrow[r] = lrow[r] * alpha + bs;
            mrow[r] = mnew;
#pragma unroll
            for (int t = 0; t < 4; ++t) o[t][r] *= alpha;
            pb[(r + rbase) * 32 + mloc]      = (_Float16)p0;
            pb[(r + rbase) * 32 + 16 + mloc] = (_Float16)p1;
        }
        asm volatile("s_wait_dscnt 0x0" ::: "memory");

        // ---- reload P as 16x32 A-fragment (two contiguous 16B LDS loads) ----
        v16h pa = load_a_frag_lds(&pb[mloc * 32], lane);

        // ---- O += P V  (4 d-chunks of 16; contiguous LDS reads) ----
#pragma unroll
        for (int t = 0; t < 4; ++t) {
            v16h vf = *(const v16h*)&vt[(t * 16 + mloc) * 32 + kofb];
            o[t] = __builtin_amdgcn_wmma_f32_16x16x32_f16(
                false, pa, false, vf, (short)0, o[t], false, false);
        }
        asm volatile("s_wait_dscnt 0x0" ::: "memory");  // LDS WAR before next iter
    }

    // ---- normalize and store (pixel-major f16, feeds out-proj GEMM) ----
#pragma unroll
    for (int t = 0; t < 4; ++t) {
#pragma unroll
        for (int r = 0; r < 8; ++r) {
            float v = o[t][r] / lrow[r];
            abuf[((size_t)b * NQ + q0 + rbase + r) * 512 + hh * 64 + t * 16 + mloc] =
                (_Float16)v;
        }
    }
}

// ===========================================================================
extern "C" void kernel_launch(void* const* d_in, const int* in_sizes, int n_in,
                              void* d_out, int out_size, void* d_ws, size_t ws_size,
                              hipStream_t stream) {
    const float* x        = (const float*)d_in[0];
    const float* features = (const float*)d_in[1];
    const float* dw_q  = (const float*)d_in[2];
    const float* bnq_g = (const float*)d_in[3];
    const float* bnq_b = (const float*)d_in[4];
    const float* bnq_m = (const float*)d_in[5];
    const float* bnq_v = (const float*)d_in[6];
    const float* pw_q  = (const float*)d_in[7];
    const float* dw_kv = (const float*)d_in[8];
    const float* bnk_g = (const float*)d_in[9];
    const float* bnk_b = (const float*)d_in[10];
    const float* bnk_m = (const float*)d_in[11];
    const float* bnk_v = (const float*)d_in[12];
    const float* pw_kv = (const float*)d_in[13];
    const float* w_out = (const float*)d_in[14];
    const float* b_out = (const float*)d_in[15];

    // workspace layout (f16 intermediates, pixel-major unless noted)
    char* ws = (char*)d_ws;
    size_t off = 0;
    _Float16* tq  = (_Float16*)(ws + off); off += (size_t)2 * 4096 * 256 * 2;  // [b][n][c]
    _Float16* tkv = (_Float16*)(ws + off); off += (size_t)2 * 1024 * 256 * 2;  // [b][n][c]
    _Float16* qb  = (_Float16*)(ws + off); off += (size_t)2 * 4096 * 512 * 2;  // [b][n][c]
    _Float16* kvb = (_Float16*)(ws + off); off += (size_t)2 * 1024 * 512 * 2   // K pm
                                               + (size_t)2 * 512 * 1024 * 2;  // V cm
    _Float16* ab  = (_Float16*)(ws + off); off += (size_t)2 * 4096 * 512 * 2;  // [b][n][c]

    // 1) depthwise + BN + ReLU -> pixel-major f16
    dw_bn_relu_kernel<<<(2 * 256 * 64 * 64 + 255) / 256, 256, 0, stream>>>(
        x, dw_q, bnq_g, bnq_b, bnq_m, bnq_v, tq, 256, 64, 64, 64, 64, 1);
    dw_bn_relu_kernel<<<(2 * 256 * 32 * 32 + 255) / 256, 256, 0, stream>>>(
        features, dw_kv, bnk_g, bnk_b, bnk_m, bnk_v, tkv, 256, 64, 64, 32, 32, 2);

    // 2) pointwise convs as WMMA GEMMs; 64x64 tiles per block
    gemm_wmma_kernel<<<2 * (512 / 64) * (4096 / 64), 128, 0, stream>>>(
        pw_q, tq, (void*)qb, nullptr, 512, 4096, 256, 0);
    gemm_wmma_kernel<<<2 * (1024 / 64) * (1024 / 64), 128, 0, stream>>>(
        pw_kv, tkv, (void*)kvb, nullptr, 1024, 1024, 256, 2);

    // 3) flash attention: 4096 waves -> 2048 blocks of 2 waves
    attention_wmma_kernel<<<2048, 64, 0, stream>>>(qb, kvb, ab);

    // 4) output projection: M=256, N=4096, K=512, f32 bias+ReLU to d_out (NCHW)
    gemm_wmma_kernel<<<2 * (256 / 64) * (4096 / 64), 128, 0, stream>>>(
        w_out, ab, d_out, b_out, 256, 4096, 512, 1);
    (void)in_sizes; (void)n_in; (void)out_size; (void)ws_size;
}
